// VectorQuantizer_49417893707872
// MI455X (gfx1250) — matrix operand order repaired and checked by reading, not compiled, not run on hardware
//
#include <hip/hip_runtime.h>
#include <hip/hip_bf16.h>

// ---------------------------------------------------------------------------
// VQ-VAE vector quantizer for MI455X (gfx1250, wave32, WMMA).
// Heavy op: argmin_k ||x_n - c_k||^2  ==  argmin_k (||c_k||^2 - 2 x_n.c_k)
// computed with bf16x3-split V_WMMA_F32_16X16X32_BF16 (near-fp32 accuracy,
// bf16-class matrix throughput). Everything else is bandwidth-bound streaming.
// ---------------------------------------------------------------------------

#define NTOK  32768     // 16*2048 tokens
#define DDIM  512
#define KC    4096
#define LDST  520       // LDS row stride in bf16 elems (512 + 8 pad -> bank stride 4)

typedef __attribute__((ext_vector_type(16))) __bf16 v16bf;
typedef __attribute__((ext_vector_type(8)))  float  v8f;

union HalfOp { v16bf v; int4 q[2]; };

// ------------------------------ small helpers ------------------------------

__global__ void zero_small_kernel(int* counts, float* loss) {
    int i = blockIdx.x * blockDim.x + threadIdx.x;
    if (i < KC) counts[i] = 0;
    if (i == 0) *loss = 0.0f;
}

__global__ void zero_fill4_kernel(float4* p, size_t n4) {
    size_t i = (size_t)blockIdx.x * blockDim.x + threadIdx.x;
    size_t stride = (size_t)gridDim.x * blockDim.x;
    float4 z = {0.f, 0.f, 0.f, 0.f};
    for (; i < n4; i += stride) p[i] = z;
}

__global__ void scale_copy_kernel(const float* __restrict__ in, float* __restrict__ out,
                                  float s, size_t n) {
    size_t i = (size_t)blockIdx.x * blockDim.x + threadIdx.x;
    size_t stride = (size_t)gridDim.x * blockDim.x;
    for (; i < n; i += stride) out[i] = s * in[i];
}

// codebook -> bf16 hi/lo planes + squared norms. 8 code rows per 256-thread block.
__global__ void prep_codebook_kernel(const float* __restrict__ cb,
                                     __bf16* __restrict__ CH, __bf16* __restrict__ CL,
                                     float* __restrict__ cnorm) {
    const int rb   = blockIdx.x * 8;
    const int tid  = threadIdx.x;
    const int lane = tid & 31;
    const int wv   = tid >> 5;

    for (int i = tid; i < 8 * DDIM; i += 256) {
        int r = i >> 9, d = i & (DDIM - 1);
        float x  = cb[(size_t)(rb + r) * DDIM + d];
        __bf16 h = (__bf16)x;
        CH[(size_t)(rb + r) * DDIM + d] = h;
        CL[(size_t)(rb + r) * DDIM + d] = (__bf16)(x - (float)h);
    }
    // one wave per code row: ||c||^2
    int r = rb + wv;
    float s = 0.f;
    for (int d = lane; d < DDIM; d += 32) {
        float x = cb[(size_t)r * DDIM + d];
        s += x * x;
    }
    #pragma unroll
    for (int off = 16; off; off >>= 1) s += __shfl_xor(s, off, 32);
    if (lane == 0) cnorm[r] = s;
}

// ------------------------- the WMMA argmin GEMM ----------------------------
// Block: 256 threads = 8 waves tiled 2(rows)x4(codes); block owns 32 token
// rows and loops over all K=4096 codes in 64-code tiles. dot = xh*ch + xh*cl
// + xl*ch accumulated in one v8f, then score = ||c||^2 - 2*dot, reduced with
// a 16-lane shuffle argmin (tie-break = lowest index, matching jnp.argmin).

__global__ void __launch_bounds__(256, 1)
argmin_gemm_kernel(const float* __restrict__ X,
                   const __bf16* __restrict__ CH, const __bf16* __restrict__ CL,
                   const float* __restrict__ cnorm,
                   int* __restrict__ idx_out, float* __restrict__ idxf_out) {
    extern __shared__ __align__(16) char smraw[];
    __bf16* sXH = (__bf16*)smraw;          // 32 x LDST
    __bf16* sXL = sXH + 32 * LDST;         // 32 x LDST
    __bf16* sCH = sXL + 32 * LDST;         // 64 x LDST
    __bf16* sCL = sCH + 64 * LDST;         // 64 x LDST

    const int tid    = threadIdx.x;
    const int lane   = tid & 31;
    const int wave   = tid >> 5;
    const int wr     = wave >> 2;          // row tile 0..1
    const int wc     = wave & 3;           // code tile 0..3
    const int lhalf  = lane >> 4;          // 0 / 1 half-wave
    const int lr     = lane & 15;

    const int rowbase = blockIdx.x * 32;

    // Stage this block's 32 X rows once, split into bf16 hi + residual.
    for (int i = tid; i < 32 * DDIM; i += 256) {
        int r = i >> 9, d = i & (DDIM - 1);
        float x  = X[(size_t)(rowbase + r) * DDIM + d];
        __bf16 h = (__bf16)x;
        sXH[r * LDST + d] = h;
        sXL[r * LDST + d] = (__bf16)(x - (float)h);
    }

    float rmin[8];
    int   ridx[8];
    #pragma unroll
    for (int i = 0; i < 8; ++i) { rmin[i] = 3.4e38f; ridx[i] = 0; }

    const int arow = wr * 16 + lr;         // X LDS row feeding A operand
    const int crow = wc * 16 + lr;         // codebook LDS row feeding B operand

    for (int kb = 0; kb < KC; kb += 64) {
        __syncthreads();   // previous tile fully consumed / X stage done
        // stage 64-code bf16 tile, both planes, 16B chunks
        for (int i = tid; i < 64 * 64; i += 256) {
            int r = i >> 6, c = (i & 63) << 3;   // 8 bf16 per int4
            *(int4*)(sCH + r * LDST + c) = *(const int4*)(CH + (size_t)(kb + r) * DDIM + c);
            *(int4*)(sCL + r * LDST + c) = *(const int4*)(CL + (size_t)(kb + r) * DDIM + c);
        }
        __syncthreads();

        v8f acc = {0.f, 0.f, 0.f, 0.f, 0.f, 0.f, 0.f, 0.f};
        #pragma unroll 4
        for (int d0 = 0; d0 < DDIM; d0 += 32) {
            HalfOp ah, al, bh, bl;
            // A operand, contiguous k order used here: lanes<16 hold K0..15, lanes>=16 hold K16..31
            const int oa0 = d0 + lhalf * 16;
            const int oa1 = oa0 + 8;
            // B operand, same contiguous k order as A: lanes<16 hold K0..15, lanes>=16 hold K16..31
            const int ob0 = d0 + lhalf * 16;
            const int ob1 = ob0 + 8;
            ah.q[0] = *(const int4*)(sXH + arow * LDST + oa0);
            ah.q[1] = *(const int4*)(sXH + arow * LDST + oa1);
            al.q[0] = *(const int4*)(sXL + arow * LDST + oa0);
            al.q[1] = *(const int4*)(sXL + arow * LDST + oa1);
            bh.q[0] = *(const int4*)(sCH + crow * LDST + ob0);
            bh.q[1] = *(const int4*)(sCH + crow * LDST + ob1);
            bl.q[0] = *(const int4*)(sCL + crow * LDST + ob0);
            bl.q[1] = *(const int4*)(sCL + crow * LDST + ob1);
            // bf16x3: xh*ch + xh*cl + xl*ch  (xl*cl ~ 2^-16, dropped)
            acc = __builtin_amdgcn_wmma_f32_16x16x32_bf16(false, ah.v, false, bh.v,
                                                          (short)0, acc, false, false);
            acc = __builtin_amdgcn_wmma_f32_16x16x32_bf16(false, ah.v, false, bl.v,
                                                          (short)0, acc, false, false);
            acc = __builtin_amdgcn_wmma_f32_16x16x32_bf16(false, al.v, false, bh.v,
                                                          (short)0, acc, false, false);
        }

        const int code0 = kb + wc * 16 + lr;   // this lane's code column
        const float cn  = cnorm[code0];
        #pragma unroll
        for (int i = 0; i < 8; ++i) {
            float s = cn - 2.0f * acc[i];
            int  ci = code0;
            #pragma unroll
            for (int off = 8; off; off >>= 1) {   // 16-lane argmin within half-wave
                float os = __shfl_xor(s, off, 32);
                int   oc = __shfl_xor(ci, off, 32);
                if (os < s || (os == s && oc < ci)) { s = os; ci = oc; }
            }
            if (s < rmin[i] || (s == rmin[i] && ci < ridx[i])) { rmin[i] = s; ridx[i] = ci; }
        }
    }

    __syncthreads();
    // combine the 4 code-tile waves per row via LDS
    float* cval = (float*)smraw;           // [4][32]
    int*   cidx = (int*)(cval + 128);      // [4][32]
    if (lr == 0) {                         // lanes 0 and 16 each own 8 rows
        #pragma unroll
        for (int i = 0; i < 8; ++i) {
            int row = wr * 16 + lhalf * 8 + i;
            cval[wc * 32 + row] = rmin[i];
            cidx[wc * 32 + row] = ridx[i];
        }
    }
    __syncthreads();
    if (tid < 32) {
        float best = cval[tid];
        int   bi   = cidx[tid];
        for (int c = 1; c < 4; ++c) {
            float v  = cval[c * 32 + tid];
            int   vi = cidx[c * 32 + tid];
            if (v < best || (v == best && vi < bi)) { best = v; bi = vi; }
        }
        int grow = rowbase + tid;
        idx_out[grow]  = bi;
        idxf_out[grow] = (float)bi;
    }
}

// ------------------------- post-argmin streaming ---------------------------

__global__ void scatter_ones_kernel(const int* __restrict__ idx, float* __restrict__ enc) {
    int n = blockIdx.x * blockDim.x + threadIdx.x;
    if (n < NTOK) enc[(size_t)n * KC + idx[n]] = 1.0f;
}

// one block per token: gather quantized row, commitment-loss partial,
// counts histogram, ema_weight scatter-add of (1-decay)*x.
__global__ void token_update_kernel(const float* __restrict__ X, const float* __restrict__ cb,
                                    const int* __restrict__ idx,
                                    float* __restrict__ quant, float* __restrict__ emaw,
                                    int* __restrict__ counts, float* __restrict__ loss) {
    const int n = blockIdx.x;
    const int k = idx[n];
    const float* xr = X  + (size_t)n * DDIM;
    const float* cr = cb + (size_t)k * DDIM;
    float part = 0.f;
    for (int d = threadIdx.x; d < DDIM; d += 256) {
        float x = xr[d], q = cr[d];
        quant[(size_t)n * DDIM + d] = x + (q - x);   // ST arithmetic as in JAX
        float df = x - q;
        part += df * df;
        atomicAdd(&emaw[(size_t)k * DDIM + d], 0.01f * x);
    }
    #pragma unroll
    for (int off = 16; off; off >>= 1) part += __shfl_xor(part, off, 32);
    __shared__ float wsum[8];
    int lane = threadIdx.x & 31, wv = threadIdx.x >> 5;
    if (lane == 0) wsum[wv] = part;
    __syncthreads();
    if (threadIdx.x == 0) {
        float t = 0.f;
        for (int i = 0; i < 8; ++i) t += wsum[i];
        atomicAdd(loss, t);
        atomicAdd(&counts[k], 1);
    }
}

__global__ void finalize_counts_kernel(const float* __restrict__ ema_count,
                                       const int* __restrict__ counts,
                                       const float* __restrict__ loss_acc,
                                       float* __restrict__ out_ema_count,
                                       float* __restrict__ out_commit,
                                       float* __restrict__ out_cbloss,
                                       float* __restrict__ inv_norm) {
    __shared__ float red[1024];
    const int tid = threadIdx.x;
    float local = 0.f;
    for (int k = tid; k < KC; k += 1024) {
        float nc = 0.99f * ema_count[k] + 0.01f * (float)counts[k];
        out_ema_count[k] = nc;
        local += nc;
    }
    red[tid] = local;
    __syncthreads();
    for (int s = 512; s; s >>= 1) {
        if (tid < s) red[tid] += red[tid + s];
        __syncthreads();
    }
    const float n = red[0];
    for (int k = tid; k < KC; k += 1024) {
        float nc   = 0.99f * ema_count[k] + 0.01f * (float)counts[k];
        float norm = (nc + 1e-5f) / (n + (float)KC * 1e-5f) * n;
        inv_norm[k] = 1.0f / norm;
    }
    if (tid == 0) {
        *out_commit = loss_acc[0] * (1.0f / (float)((size_t)NTOK * DDIM));
        *out_cbloss = 0.0f;
    }
}

__global__ void finalize_codebook_kernel(const float* __restrict__ emaw,
                                         const float* __restrict__ inv_norm,
                                         float* __restrict__ newcb) {
    size_t i = (size_t)blockIdx.x * blockDim.x + threadIdx.x;
    size_t stride = (size_t)gridDim.x * blockDim.x;
    const size_t n = (size_t)KC * DDIM;
    for (; i < n; i += stride) newcb[i] = emaw[i] * inv_norm[i >> 9];
}

// ------------------------------- launcher ----------------------------------

extern "C" void kernel_launch(void* const* d_in, const int* in_sizes, int n_in,
                              void* d_out, int out_size, void* d_ws, size_t ws_size,
                              hipStream_t stream) {
    const float* X         = (const float*)d_in[0];   // [16,2048,512]
    const float* codebook  = (const float*)d_in[1];   // [4096,512]
    const float* ema_count = (const float*)d_in[2];   // [4096]
    const float* ema_weight= (const float*)d_in[3];   // [4096,512]
    float* out = (float*)d_out;

    // output layout (flat f32, reference tuple order)
    const size_t O_QUANT  = 0;                         // 16,777,216
    const size_t O_ENC    = 16777216;                  // 134,217,728
    const size_t O_IDX    = 150994944;                 // 32,768
    const size_t O_COMMIT = 151027712;
    const size_t O_CBLOSS = 151027713;
    const size_t O_NEWCB  = 151027714;                 // 2,097,152
    const size_t O_EMACNT = 153124866;                 // 4,096
    const size_t O_EMAW   = 153128962;                 // 2,097,152

    // workspace layout (bytes), ~8.6 MB total
    char* ws = (char*)d_ws;
    __bf16* CH     = (__bf16*)(ws + 0);                // 4,194,304
    __bf16* CL     = (__bf16*)(ws + 4194304);          // 4,194,304
    float*  cnorm  = (float*) (ws + 8388608);          // 16,384
    int*    idx    = (int*)   (ws + 8404992);          // 131,072
    int*    counts = (int*)   (ws + 8536064);          // 16,384
    float*  invn   = (float*) (ws + 8552448);          // 16,384
    float*  lossa  = (float*) (ws + 8568832);          // 4

    const int SMEM = (32 + 32 + 64 + 64) * LDST * 2;   // 199,680 B dynamic LDS
    hipFuncSetAttribute(reinterpret_cast<const void*>(argmin_gemm_kernel),
                        hipFuncAttributeMaxDynamicSharedMemorySize, SMEM);

    // 1. zero scratch accumulators
    zero_small_kernel<<<16, 256, 0, stream>>>(counts, lossa);
    // 2. codebook -> bf16 hi/lo + norms (L2-resident afterwards)
    prep_codebook_kernel<<<KC / 8, 256, 0, stream>>>(codebook, CH, CL, cnorm);
    // 3. zero the 512 MB one-hot output
    zero_fill4_kernel<<<8192, 256, 0, stream>>>((float4*)(out + O_ENC),
                                                (size_t)NTOK * KC / 4);
    // 4. WMMA argmin over all codes
    argmin_gemm_kernel<<<NTOK / 32, 256, SMEM, stream>>>(X, CH, CL, cnorm,
                                                         idx, out + O_IDX);
    // 5. one-hot scatter
    scatter_ones_kernel<<<NTOK / 256, 256, 0, stream>>>(idx, out + O_ENC);
    // 6. new_ema_weight = decay * ema_weight (then scatter-added in 7)
    scale_copy_kernel<<<2048, 256, 0, stream>>>(ema_weight, out + O_EMAW, 0.99f,
                                                (size_t)KC * DDIM);
    // 7. gather quantized, commitment loss, counts, ema_weight adds
    token_update_kernel<<<NTOK, 256, 0, stream>>>(X, codebook, idx, out + O_QUANT,
                                                  out + O_EMAW, counts, lossa);
    // 8. EMA counts, normalization, scalar losses
    finalize_counts_kernel<<<1, 1024, 0, stream>>>(ema_count, counts, lossa,
                                                   out + O_EMACNT, out + O_COMMIT,
                                                   out + O_CBLOSS, invn);
    // 9. new_codebook = new_ema_weight / normalized_count
    finalize_codebook_kernel<<<2048, 256, 0, stream>>>(out + O_EMAW, invn, out + O_NEWCB);

    (void)in_sizes; (void)n_in; (void)out_size; (void)ws_size;
}
